// MessagePassingLayer_22840636080227
// MI455X (gfx1250) — matrix operand order repaired and checked by reading, not compiled, not run on hardware
//
#include <hip/hip_runtime.h>
#include <hip/hip_bf16.h>

#define N_NODES 2048
#define D_DIM   256
#define DH_DIM  256
#define H_HEADS 4
#define DHEAD   64

typedef __attribute__((ext_vector_type(16))) _Float16 v16h;
typedef __attribute__((ext_vector_type(8)))  _Float16 v8h;
typedef __attribute__((ext_vector_type(8)))  float    v8f;

__device__ __forceinline__ v16h pack16(v8h lo, v8h hi) {
  v16h r;
#pragma unroll
  for (int e = 0; e < 8; e++) { r[e] = lo[e]; r[8 + e] = hi[e]; }
  return r;
}

// Async Global -> LDS copy, 16B per lane (GV mode: LDS-offset VGPR, 64b vaddr)
__device__ __forceinline__ void async_load_b128(unsigned lds_off, const void* gaddr) {
  asm volatile("global_load_async_to_lds_b128 %0, %1, off"
               :: "v"(lds_off), "v"(gaddr) : "memory");
}

template <int N>
__device__ __forceinline__ void wait_asynccnt() {
#if defined(__has_builtin) && __has_builtin(__builtin_amdgcn_s_wait_asynccnt)
  __builtin_amdgcn_s_wait_asynccnt(N);
#else
  asm volatile("s_wait_asynccnt %0" :: "i"(N) : "memory");
#endif
}

// ---------------------------------------------------------------------------
// f32 -> f16 conversion
// ---------------------------------------------------------------------------
__global__ void f32_to_f16_kernel(const float* __restrict__ in,
                                  _Float16* __restrict__ out, int n) {
  int i = blockIdx.x * blockDim.x + threadIdx.x;
  if (i < n) out[i] = (_Float16)in[i];
}

// ---------------------------------------------------------------------------
// a_cur / a_nb projections: x @ Wa[:, :D].T and x @ Wa[:, D:].T  -> [N, H]
// ---------------------------------------------------------------------------
__global__ void attn_coef_kernel(const float* __restrict__ x,
                                 const float* __restrict__ Wa0,
                                 const float* __restrict__ Wa1,
                                 float* __restrict__ a0c, float* __restrict__ a0n,
                                 float* __restrict__ a1c, float* __restrict__ a1n) {
  int t = blockIdx.x * blockDim.x + threadIdx.x;
  if (t >= N_NODES * 16) return;
  int node = t >> 4, slot = t & 15;
  int layer = slot >> 3, half = (slot >> 2) & 1, h = slot & 3;
  const float* Wa = layer ? Wa1 : Wa0;
  const float* wrow = Wa + h * (2 * D_DIM) + half * D_DIM;
  const float* xr = x + (size_t)node * D_DIM;
  float acc = 0.f;
#pragma unroll 4
  for (int k = 0; k < D_DIM; k++) acc += xr[k] * wrow[k];
  float* out = layer ? (half ? a1n : a1c) : (half ? a0n : a0c);
  out[node * H_HEADS + h] = acc;
}

// ---------------------------------------------------------------------------
// Generic WMMA f16 GEMM: C[M,Nc] = A[M,K] @ W[Nc,K]^T + bias
// block = 128 threads (4 waves); wave -> 16(M) x 64(Nc) tile; K step 32
// ---------------------------------------------------------------------------
__global__ void gemm_wmma_kernel(const _Float16* __restrict__ A,
                                 const _Float16* __restrict__ W,
                                 const float* __restrict__ bias,
                                 int M, int Nc, int K,
                                 float* __restrict__ outF,
                                 _Float16* __restrict__ outT,
                                 _Float16* __restrict__ outH) {
  const int lane = threadIdx.x & 31;
  const int wv   = threadIdx.x >> 5;
  const int hh   = lane >> 4;
  const int ln   = lane & 15;
  const int i0   = blockIdx.x * 64 + wv * 16;
  const int n0   = blockIdx.y * 64;

  v8f acc[4] = {};
  const _Float16* arow = A + (size_t)(i0 + ln) * K;

  for (int kb = 0; kb < K; kb += 32) {
    v16h Aop = pack16(*(const v8h*)(arow + kb + hh * 8),
                      *(const v8h*)(arow + kb + 16 + hh * 8));
#pragma unroll
    for (int t = 0; t < 4; t++) {
      const _Float16* brow = W + (size_t)(n0 + t * 16 + ln) * K + kb + hh * 8;
      v16h Bop = pack16(*(const v8h*)brow, *(const v8h*)(brow + 16));
      acc[t] = __builtin_amdgcn_wmma_f32_16x16x32_f16(
          false, Aop, false, Bop, (short)0, acc[t], false, false);
    }
  }

#pragma unroll
  for (int t = 0; t < 4; t++) {
    int col = n0 + t * 16 + ln;
    float b = bias ? bias[col] : 0.f;
#pragma unroll
    for (int r = 0; r < 8; r++) {
      int orow = i0 + r + 8 * hh;
      float v = acc[t][r] + b;
      if (outF) outF[(size_t)orow * Nc + col] = v;
      if (outT) outT[(size_t)col * M + orow] = (_Float16)v;
      if (outH) outH[(size_t)orow * Nc + col] = (_Float16)v;
    }
  }
}

// ---------------------------------------------------------------------------
// Fused masked-softmax attention aggregation (flash-style, single pass).
// grid = (N/16 i-tiles, 2 layers); block = 128 threads, wave = head.
//
// The HBM-dominant adj/w streams (67 MB total) are staged into LDS ONCE per
// block (shared by all 4 head-waves) with double-buffered async b128 copies
// (global_load_async_to_lds_b128 + s_wait_asynccnt), overlapping the next
// chunk's fetch with the current chunk's exp/WMMA work. LDS rows use a
// 20-float pitch so the two half-waves read disjoint bank groups.
// ---------------------------------------------------------------------------
#define SPITCH 20  // floats per staged row (32 rows x 16 cols, padded)

__global__ void attn_agg_kernel(const float* __restrict__ adj0, const float* __restrict__ w0,
                                const float* __restrict__ adj1, const float* __restrict__ w1,
                                const float* __restrict__ a0c, const float* __restrict__ a0n,
                                const float* __restrict__ a1c, const float* __restrict__ a1n,
                                const float* __restrict__ ba0, const float* __restrict__ ba1,
                                const _Float16* __restrict__ msgT0,
                                const _Float16* __restrict__ msgT1,
                                _Float16* __restrict__ mcat /* [N, 2*DH] f16 */) {
  __shared__ float sW[2][32 * SPITCH];
  __shared__ float sA[2][32 * SPITCH];

  const int tid  = threadIdx.x;
  const int lane = tid & 31;
  const int h    = tid >> 5;   // wave == head
  const int hh   = lane >> 4;
  const int ln   = lane & 15;
  const int i0   = blockIdx.x * 16;
  const int layer = blockIdx.y;

  const float* adj = layer ? adj1 : adj0;
  const float* wgt = layer ? w1 : w0;
  const float* ac  = layer ? a1c : a0c;
  const float* an  = layer ? a1n : a0n;
  const float  bah = (layer ? ba1 : ba0)[h];
  const _Float16* msgT = layer ? msgT1 : msgT0;

  const int   ig   = i0 + ln;
  const float acur = ac[ig * H_HEADS + h] + bah;

  // per-thread async-copy assignment: row = j within chunk, seg = 16B segment
  const int arow_ = tid >> 2;   // 0..31
  const int aseg_ = tid & 3;    // 0..3
  const unsigned swBase = (unsigned)(size_t)&sW[0][0];
  const unsigned saBase = (unsigned)(size_t)&sA[0][0];
  const unsigned bufStride = (unsigned)(32 * SPITCH * sizeof(float));
  const unsigned ldsRow = (unsigned)(arow_ * SPITCH * sizeof(float) + aseg_ * 16);

  auto issue_chunk = [&](int j0, int b) {
    const size_t goff = (size_t)(j0 + arow_) * N_NODES + i0 + aseg_ * 4;
    async_load_b128(swBase + (unsigned)b * bufStride + ldsRow, wgt + goff);
    async_load_b128(saBase + (unsigned)b * bufStride + ldsRow, adj + goff);
  };

  v8f acc[4] = {};
  float lsum = 0.f;

  issue_chunk(0, 0);  // prologue

  const int NCHUNK = N_NODES / 32;
  for (int c = 0; c < NCHUNK; c++) {
    const int b = c & 1;
    const int j0 = c * 32;
    if (c + 1 < NCHUNK) {
      issue_chunk(j0 + 32, (c + 1) & 1);
      wait_asynccnt<2>();   // in-order completion: prev chunk's pair retired
    } else {
      wait_asynccnt<0>();
    }
    __syncthreads();        // all waves' staged data visible

    // Build A operand (P^T tile, row i = ln): element e <-> K per ISA layout
    v16h Aop;
#pragma unroll
    for (int e = 0; e < 16; e++) {
      int K = (e < 8) ? (hh * 8 + e) : (16 + hh * 8 + (e - 8));
      int j = j0 + K;
      float s = an[j * H_HEADS + h] + acur;          // raw score (broadcast ld)
      s = (s > 0.f) ? s : 0.2f * s;                  // leaky_relu(0.2)
      s *= sW[b][K * SPITCH + ln];                   // edge-weight scaling (LDS)
      float m = sA[b][K * SPITCH + ln];              // mask (LDS)
      float p = (m != 0.f) ? __expf(s) : 0.f;        // masked exp
      lsum += p;
      Aop[e] = (_Float16)p;
    }
    // 4 WMMAs: agg[i, d-tile] += P^T @ msg ; B from transposed msg => b128s
#pragma unroll
    for (int t = 0; t < 4; t++) {
      const _Float16* bp =
          msgT + (size_t)(h * DHEAD + t * 16 + ln) * N_NODES + j0 + hh * 8;
      v16h Bop = pack16(*(const v8h*)bp, *(const v8h*)(bp + 16));
      acc[t] = __builtin_amdgcn_wmma_f32_16x16x32_f16(
          false, Aop, false, Bop, (short)0, acc[t], false, false);
    }
    __syncthreads();        // buffer b fully consumed before it is re-filled
  }

  // combine half-wave partial row sums; lane L ends with total for row L&15
  lsum += __shfl_xor(lsum, 16, 32);
  float lrow[8];
#pragma unroll
  for (int r = 0; r < 8; r++) lrow[r] = __shfl(lsum, r + 8 * hh, 32);

#pragma unroll
  for (int t = 0; t < 4; t++) {
    int col = layer * DH_DIM + h * DHEAD + t * 16 + ln;
#pragma unroll
    for (int r = 0; r < 8; r++) {
      int orow = i0 + r + 8 * hh;
      float v = (lrow[r] > 0.f) ? acc[t][r] / lrow[r] : 0.f;  // isolated -> 0
      mcat[(size_t)orow * (2 * DH_DIM) + col] = (_Float16)v;
    }
  }
}

// ---------------------------------------------------------------------------
// Fused GRU cell + LayerNorm: one 256-thread block per node
// ---------------------------------------------------------------------------
__global__ void gru_ln_kernel(const float* __restrict__ gi, const float* __restrict__ gh,
                              const float* __restrict__ x,
                              const float* __restrict__ g, const float* __restrict__ b,
                              float* __restrict__ out) {
  const int i = blockIdx.x;
  const int d = threadIdx.x;
  const float* gir = gi + (size_t)i * 768;
  const float* ghr = gh + (size_t)i * 768;
  float r = 1.f / (1.f + __expf(-(gir[d] + ghr[d])));
  float z = 1.f / (1.f + __expf(-(gir[256 + d] + ghr[256 + d])));
  float n = tanhf(gir[512 + d] + r * ghr[512 + d]);
  float hv = (1.f - z) * n + z * x[(size_t)i * 256 + d];

  __shared__ float s1[256], s2[256];
  s1[d] = hv; s2[d] = hv * hv;
  __syncthreads();
  for (int off = 128; off > 0; off >>= 1) {
    if (d < off) { s1[d] += s1[d + off]; s2[d] += s2[d + off]; }
    __syncthreads();
  }
  float mu  = s1[0] * (1.f / 256.f);
  float var = s2[0] * (1.f / 256.f) - mu * mu;
  float inv = rsqrtf(var + 1e-5f);
  out[(size_t)i * 256 + d] = (hv - mu) * inv * g[d] + b[d];
}

// ---------------------------------------------------------------------------
extern "C" void kernel_launch(void* const* d_in, const int* in_sizes, int n_in,
                              void* d_out, int out_size, void* d_ws, size_t ws_size,
                              hipStream_t stream) {
  const float* x    = (const float*)d_in[0];
  const float* adj0 = (const float*)d_in[1];
  const float* adj1 = (const float*)d_in[2];
  const float* w0   = (const float*)d_in[3];
  const float* w1   = (const float*)d_in[4];
  const float* Wm0  = (const float*)d_in[5];
  const float* bm0  = (const float*)d_in[6];
  const float* Wa0  = (const float*)d_in[7];
  const float* ba0  = (const float*)d_in[8];
  const float* Wm1  = (const float*)d_in[9];
  const float* bm1  = (const float*)d_in[10];
  const float* Wa1  = (const float*)d_in[11];
  const float* ba1  = (const float*)d_in[12];
  const float* wih  = (const float*)d_in[13];
  const float* whh  = (const float*)d_in[14];
  const float* bih  = (const float*)d_in[15];
  const float* bhh  = (const float*)d_in[16];
  const float* lng  = (const float*)d_in[17];
  const float* lnb  = (const float*)d_in[18];
  float* out = (float*)d_out;

  char* ws = (char*)d_ws;
  size_t off = 0;
  auto alloc = [&](size_t bytes) -> char* {
    char* p = ws + off;
    off = (off + bytes + 255) & ~(size_t)255;
    return p;
  };
  _Float16* xh    = (_Float16*)alloc((size_t)N_NODES * D_DIM * 2);
  _Float16* Wm0h  = (_Float16*)alloc((size_t)DH_DIM * D_DIM * 2);
  _Float16* Wm1h  = (_Float16*)alloc((size_t)DH_DIM * D_DIM * 2);
  _Float16* whhh  = (_Float16*)alloc((size_t)768 * 256 * 2);
  _Float16* wihh  = (_Float16*)alloc((size_t)768 * 512 * 2);
  float*    a0c   = (float*)alloc((size_t)N_NODES * H_HEADS * 4);
  float*    a0n   = (float*)alloc((size_t)N_NODES * H_HEADS * 4);
  float*    a1c   = (float*)alloc((size_t)N_NODES * H_HEADS * 4);
  float*    a1n   = (float*)alloc((size_t)N_NODES * H_HEADS * 4);
  _Float16* msgT0 = (_Float16*)alloc((size_t)DH_DIM * N_NODES * 2);
  _Float16* msgT1 = (_Float16*)alloc((size_t)DH_DIM * N_NODES * 2);
  _Float16* mcat  = (_Float16*)alloc((size_t)N_NODES * 2 * DH_DIM * 2);
  float*    ghb   = (float*)alloc((size_t)N_NODES * 768 * 4);
  float*    gib   = (float*)alloc((size_t)N_NODES * 768 * 4);

  auto cvt = [&](const float* in, _Float16* o, int n) {
    f32_to_f16_kernel<<<(n + 255) / 256, 256, 0, stream>>>(in, o, n);
  };
  cvt(x,   xh,   N_NODES * D_DIM);
  cvt(Wm0, Wm0h, DH_DIM * D_DIM);
  cvt(Wm1, Wm1h, DH_DIM * D_DIM);
  cvt(whh, whhh, 768 * 256);
  cvt(wih, wihh, 768 * 512);

  attn_coef_kernel<<<(N_NODES * 16 + 255) / 256, 256, 0, stream>>>(
      x, Wa0, Wa1, a0c, a0n, a1c, a1n);

  // msg_l = x @ Wm_l^T + bm_l  -> stored transposed f16 [256][2048]
  gemm_wmma_kernel<<<dim3(N_NODES / 64, DH_DIM / 64), 128, 0, stream>>>(
      xh, Wm0h, bm0, N_NODES, DH_DIM, D_DIM, nullptr, msgT0, nullptr);
  gemm_wmma_kernel<<<dim3(N_NODES / 64, DH_DIM / 64), 128, 0, stream>>>(
      xh, Wm1h, bm1, N_NODES, DH_DIM, D_DIM, nullptr, msgT1, nullptr);
  // gh = x @ Whh^T + bhh  -> f32 [2048,768]
  gemm_wmma_kernel<<<dim3(N_NODES / 64, 768 / 64), 128, 0, stream>>>(
      xh, whhh, bhh, N_NODES, 768, 256, ghb, nullptr, nullptr);

  // fused attention aggregation for both layers -> mcat f16 [2048, 512]
  attn_agg_kernel<<<dim3(N_NODES / 16, 2), 128, 0, stream>>>(
      adj0, w0, adj1, w1, a0c, a0n, a1c, a1n, ba0, ba1, msgT0, msgT1, mcat);

  // gi = mcat @ Wih^T + bih -> f32 [2048,768]
  gemm_wmma_kernel<<<dim3(N_NODES / 64, 768 / 64), 128, 0, stream>>>(
      mcat, wihh, bih, N_NODES, 768, 512, gib, nullptr, nullptr);

  // GRU + LayerNorm
  gru_ln_kernel<<<N_NODES, 256, 0, stream>>>(gib, ghb, x, lng, lnb, out);
}
